// Attention_44032004718769
// MI455X (gfx1250) — compile-verified, matrix operand order checked by reading, not run on hardware
//
#include <hip/hip_runtime.h>
#include <hip/hip_bf16.h>

// MI455X (gfx1250) multi-head attention, f16 WMMA with f32 accumulation.
// EMBED=HIDDEN=OUT=1024, H=16 heads, head_dim=64, B=2, P=C=2048.
// Reference reshape puts head index as the FASTEST axis (c = d*16 + h),
// and the mask is added AFTER softmax -> two-pass softmax (m,l) scheme.

typedef __attribute__((ext_vector_type(16))) _Float16 v16h;
typedef __attribute__((ext_vector_type(8)))  float    v8f;

#define NH   16      // heads
#define HD   64      // head dim / head out
#define SEQ  2048    // P == C
#define EMB  1024

union HalfVec { v16h h; uint4 u[2]; };
union Pack8   { _Float16 h[8]; uint4 u; };

// Load one 16-lane-row WMMA fragment (A or B operand, 16x32 / 32x16 f16).
// ISA layout: lane r = lane&15, k-base = (lane>>4)*8, elements 0..7 at k=kb,
// elements 8..15 at k=kb+16 -> two contiguous 16-byte chunks per lane.
__device__ __forceinline__ v16h frag_ld(const _Float16* tile, int ld, int lane) {
  const int r  = lane & 15;
  const int kb = (lane >> 4) << 3;
  HalfVec f;
  f.u[0] = *reinterpret_cast<const uint4*>(tile + (size_t)r * ld + kb);
  f.u[1] = *reinterpret_cast<const uint4*>(tile + (size_t)r * ld + kb + 16);
  return f.h;
}

// ---------------------------------------------------------------------------
// Projection GEMM: Y = (X @ W + bias) * scale, fp32 in, f16 out, repacked
// per-head. Workgroup = 256 threads (8 waves), 64x64 output tile, K-chunks of
// 32 staged in LDS (X converted to f16; W staged transposed so B-fragments
// are contiguous ds_load_b128). Each wave owns 1 row-tile x 2 col-tiles:
// A-fragment reused across 2 WMMAs per chunk.
// ---------------------------------------------------------------------------
__global__ __launch_bounds__(256)
void proj_kernel(const float* __restrict__ X, const float* __restrict__ W,
                 const float* __restrict__ bias, _Float16* __restrict__ Y,
                 float scale, int transposed) {
  __shared__ _Float16 sA[64 * 32];   // [row][k]
  __shared__ _Float16 sB[64 * 32];   // [n][k] (W tile transposed)

  const int tid  = threadIdx.x;
  const int lane = tid & 31;
  const int wave = tid >> 5;
  const int mt = wave & 3;           // row tile 0..3
  const int np = wave >> 2;          // col-pair 0..1 -> tiles np*2, np*2+1
  const int rowBase = blockIdx.y * 64;
  const int colBase = blockIdx.x * 64;

  const int ar = tid >> 2;           // A staging: row 0..63
  const int ac = (tid & 3) * 8;      //            k chunk
  const int wk = tid >> 3;           // W staging: k row 0..31
  const int wn = (tid & 7) * 8;      //            n chunk

  v8f acc[2] = { {}, {} };

  for (int kk = 0; kk < EMB; kk += 32) {
    { // stage X tile (64 rows x 32 k), fp32 -> f16
      const float* src = X + (size_t)(rowBase + ar) * EMB + kk + ac;
      float4 f0 = *reinterpret_cast<const float4*>(src);
      float4 f1 = *reinterpret_cast<const float4*>(src + 4);
      Pack8 p;
      p.h[0]=(_Float16)f0.x; p.h[1]=(_Float16)f0.y; p.h[2]=(_Float16)f0.z; p.h[3]=(_Float16)f0.w;
      p.h[4]=(_Float16)f1.x; p.h[5]=(_Float16)f1.y; p.h[6]=(_Float16)f1.z; p.h[7]=(_Float16)f1.w;
      *reinterpret_cast<uint4*>(&sA[ar * 32 + ac]) = p.u;
    }
    { // stage W tile (32 k x 64 n) transposed: sB[n][k]
      const float* src = W + (size_t)(kk + wk) * EMB + colBase + wn;
      float4 f0 = *reinterpret_cast<const float4*>(src);
      float4 f1 = *reinterpret_cast<const float4*>(src + 4);
      float f8[8] = { f0.x, f0.y, f0.z, f0.w, f1.x, f1.y, f1.z, f1.w };
#pragma unroll
      for (int e = 0; e < 8; ++e) sB[(wn + e) * 32 + wk] = (_Float16)f8[e];
    }
    __syncthreads();
    const v16h a = frag_ld(sA + mt * 16 * 32, 32, lane);
#pragma unroll
    for (int q = 0; q < 2; ++q) {
      const v16h b = frag_ld(sB + (np * 2 + q) * 16 * 32, 32, lane);
      acc[q] = __builtin_amdgcn_wmma_f32_16x16x32_f16(false, a, false, b,
                                                      (short)0, acc[q], false, false);
    }
    __syncthreads();
  }

  // Epilogue: bias, scale, f16 convert, per-head repack (c = d*16 + h).
  const int n  = lane & 15;
  const int m0 = (lane >> 4) * 8;
  const int r0 = rowBase + mt * 16 + m0;    // global row (g = 0)
  const int b  = r0 >> 11;                  // batch (rows-per-batch = 2048)
  const int i0 = r0 & (SEQ - 1);

#pragma unroll
  for (int q = 0; q < 2; ++q) {
    const int c = colBase + (np * 2 + q) * 16 + n;   // global output column
    const int h = c & 15;
    const int d = c >> 4;
    const float bv = bias[c];
    if (!transposed) {
      // Q/K: Y[((b*16+h)*2048 + i)*64 + d] (contiguous over d -> b128 frag loads)
      _Float16* dst = Y + (((size_t)b * NH + h) * SEQ) * HD + d;
#pragma unroll
      for (int g = 0; g < 8; ++g)
        dst[(size_t)(i0 + g) * HD] = (_Float16)((acc[q][g] + bv) * scale);
    } else {
      // V transposed: Y[((b*16+h)*64 + o)*2048 + j] -> 8 contiguous j = one b128
      Pack8 p;
#pragma unroll
      for (int g = 0; g < 8; ++g) p.h[g] = (_Float16)((acc[q][g] + bv) * scale);
      _Float16* dst = Y + (((size_t)b * NH + h) * HD + d) * SEQ + i0;
      *reinterpret_cast<uint4*>(dst) = p.u;
    }
  }
}

// ---------------------------------------------------------------------------
// Pass 1: per-row softmax stats (m = rowmax, l = sum exp(s-m)) per (b,h,itile)
// via online update over all 128 j-tiles. One wave per 16-row tile.
// ---------------------------------------------------------------------------
__global__ __launch_bounds__(128)
void attn_pass1(const _Float16* __restrict__ Q, const _Float16* __restrict__ K,
                float* __restrict__ mOut, float* __restrict__ lOut) {
  const int lane = threadIdx.x & 31;
  const int wid  = blockIdx.x * 4 + (threadIdx.x >> 5);
  const int it = wid & 127;
  const int h  = (wid >> 7) & 15;
  const int b  = wid >> 11;

  const _Float16* qb = Q + (((size_t)b * NH + h) * SEQ + it * 16) * HD;
  const v16h qa0 = frag_ld(qb,      HD, lane);   // d = 0..31
  const v16h qa1 = frag_ld(qb + 32, HD, lane);   // d = 32..63

  float mrun[8], lrun[8];
#pragma unroll
  for (int g = 0; g < 8; ++g) { mrun[g] = -1e30f; lrun[g] = 0.f; }

  const _Float16* kb = K + (((size_t)b * NH + h) * SEQ) * HD;
  for (int jt = 0; jt < SEQ / 16; ++jt) {
    const _Float16* kt = kb + (size_t)jt * 16 * HD;
    // prefetch next K tile (lane-spread: 32 lanes x 128B cover the next 4KB, two 2KB tiles)
    __builtin_prefetch(kt + 16 * HD + lane * 64, 0, 0);
    v8f s = {};
    s = __builtin_amdgcn_wmma_f32_16x16x32_f16(false, qa0, false,
          frag_ld(kt,      HD, lane), (short)0, s, false, false);
    s = __builtin_amdgcn_wmma_f32_16x16x32_f16(false, qa1, false,
          frag_ld(kt + 32, HD, lane), (short)0, s, false, false);

#pragma unroll
    for (int g = 0; g < 8; ++g) {
      float t = s[g];                       // row max across 16 column-lanes
#pragma unroll
      for (int off = 1; off < 16; off <<= 1) t = fmaxf(t, __shfl_xor(t, off, 32));
      const float mn = fmaxf(mrun[g], t);
      float p = __expf(s[g] - mn);          // row sum across 16 column-lanes
#pragma unroll
      for (int off = 1; off < 16; off <<= 1) p += __shfl_xor(p, off, 32);
      lrun[g] = lrun[g] * __expf(mrun[g] - mn) + p;
      mrun[g] = mn;
    }
  }

  if ((lane & 15) == 0) {
    const int r0 = it * 16 + (lane >> 4) * 8;
    float* mp = mOut + ((size_t)b * NH + h) * SEQ + r0;
    float* lp = lOut + ((size_t)b * NH + h) * SEQ + r0;
#pragma unroll
    for (int g = 0; g < 8; ++g) { mp[g] = mrun[g]; lp[g] = lrun[g]; }
  }
}

// ---------------------------------------------------------------------------
// Pass 2: recompute scores, p = exp(s-m)/l + mask, transpose P via per-wave
// LDS (C-layout -> A-layout), O += P @ V. Each wave handles 32 query rows
// (two i-subtiles) so every K B-fragment and V B-fragment is reused by two
// WMMAs. V is pre-transposed so its B-fragments are contiguous b128 loads.
// ---------------------------------------------------------------------------
__global__ __launch_bounds__(128)
void attn_pass2(const _Float16* __restrict__ Q, const _Float16* __restrict__ K,
                const _Float16* __restrict__ VT, const float* __restrict__ mIn,
                const float* __restrict__ lIn, const float* __restrict__ mask,
                float* __restrict__ out) {
  __shared__ _Float16 sP[4][2 * 16 * 32];    // per-wave P staging (2 KB each)
  const int lane = threadIdx.x & 31;
  const int wv   = threadIdx.x >> 5;
  const int wid  = blockIdx.x * 4 + wv;      // 32-row tile id
  const int it2 = wid & 63;
  const int h   = (wid >> 6) & 15;
  const int b   = wid >> 10;
  const int iBase = it2 * 32;

  const int n  = lane & 15;
  const int m0 = (lane >> 4) * 8;

  const _Float16* qb = Q + (((size_t)b * NH + h) * SEQ + iBase) * HD;
  v16h qa[2][2];
  float mrow[2][8], linv[2][8];
  const float* mkb[2];
#pragma unroll
  for (int s = 0; s < 2; ++s) {
    qa[s][0] = frag_ld(qb + (size_t)s * 16 * HD,      HD, lane);
    qa[s][1] = frag_ld(qb + (size_t)s * 16 * HD + 32, HD, lane);
    const float* mp = mIn + ((size_t)b * NH + h) * SEQ + iBase + s * 16 + m0;
    const float* lp = lIn + ((size_t)b * NH + h) * SEQ + iBase + s * 16 + m0;
#pragma unroll
    for (int g = 0; g < 8; ++g) { mrow[s][g] = mp[g]; linv[s][g] = 1.0f / lp[g]; }
    mkb[s] = mask + ((size_t)b * SEQ + iBase + s * 16 + m0) * SEQ + n;
  }

  v8f o[2][4] = { { {}, {}, {}, {} }, { {}, {}, {}, {} } };

  const _Float16* kbase = K  + (((size_t)b * NH + h) * SEQ) * HD;
  const _Float16* vbase = VT + (((size_t)b * NH + h) * HD) * SEQ;
  _Float16* myP = sP[wv];

  for (int jc = 0; jc < SEQ / 32; ++jc) {
    // prefetch next K chunk (32 lanes x 128B cover the next 4KB)
    __builtin_prefetch(kbase + (size_t)(jc + 1) * 32 * HD + lane * 64, 0, 0);
#pragma unroll
    for (int sub = 0; sub < 2; ++sub) {
      const int jt = jc * 2 + sub;
      const _Float16* kt = kbase + (size_t)jt * 16 * HD;
      const v16h b0 = frag_ld(kt,      HD, lane);
      const v16h b1 = frag_ld(kt + 32, HD, lane);
#pragma unroll
      for (int s = 0; s < 2; ++s) {          // K fragments reused for both i-subtiles
        v8f sc = {};
        sc = __builtin_amdgcn_wmma_f32_16x16x32_f16(false, qa[s][0], false, b0,
                                                    (short)0, sc, false, false);
        sc = __builtin_amdgcn_wmma_f32_16x16x32_f16(false, qa[s][1], false, b1,
                                                    (short)0, sc, false, false);
        const float* mk = mkb[s] + jt * 16;
#pragma unroll
        for (int g = 0; g < 8; ++g) {
          // normalized softmax prob + mask (mask added AFTER softmax, as ref)
          float p = __expf(sc[g] - mrow[s][g]) * linv[s][g] + mk[(size_t)g * SEQ];
          myP[s * 512 + (m0 + g) * 32 + sub * 16 + n] = (_Float16)p;
        }
      }
    }
    // Same-wave LDS RAW: DS ops are in-order, but force the wait + block
    // compiler reordering across the transpose.
    asm volatile("s_wait_dscnt 0" ::: "memory");
    __builtin_amdgcn_wave_barrier();
    const v16h pa0 = frag_ld(myP,       32, lane);   // P tiles as A-fragments
    const v16h pa1 = frag_ld(myP + 512, 32, lane);
    const _Float16* vt = vbase + jc * 32;
#pragma unroll
    for (int t = 0; t < 4; ++t) {            // V fragment reused for both i-subtiles
      const v16h vb = frag_ld(vt + (size_t)t * 16 * SEQ, SEQ, lane);
      o[0][t] = __builtin_amdgcn_wmma_f32_16x16x32_f16(false, pa0, false, vb,
                                                       (short)0, o[0][t], false, false);
      o[1][t] = __builtin_amdgcn_wmma_f32_16x16x32_f16(false, pa1, false, vb,
                                                       (short)0, o[1][t], false, false);
    }
    __builtin_amdgcn_wave_barrier();
  }

  // out[b][i][o*16 + h], fp32
#pragma unroll
  for (int s = 0; s < 2; ++s) {
    float* ob = out + ((size_t)b * SEQ + iBase + s * 16 + m0) * 1024 + h;
#pragma unroll
    for (int t = 0; t < 4; ++t)
#pragma unroll
      for (int g = 0; g < 8; ++g)
        ob[(size_t)g * 1024 + (t * 16 + n) * 16] = o[s][t][g];
  }
}

extern "C" void kernel_launch(void* const* d_in, const int* in_sizes, int n_in,
                              void* d_out, int out_size, void* d_ws, size_t ws_size,
                              hipStream_t stream) {
  (void)in_sizes; (void)n_in; (void)out_size; (void)ws_size;
  const float* problem = (const float*)d_in[0];
  const float* context = (const float*)d_in[1];
  const float* maskp   = (const float*)d_in[2];
  const float* Wq = (const float*)d_in[3];
  const float* bq = (const float*)d_in[4];
  const float* Wk = (const float*)d_in[5];
  const float* bk = (const float*)d_in[6];
  const float* Wv = (const float*)d_in[7];
  const float* bv = (const float*)d_in[8];
  float* out = (float*)d_out;

  // Workspace layout: Q16 | K16 | VT16 (8 MB each, f16) | m | l (256 KB each)
  const size_t NHALF = (size_t)2 * NH * SEQ * HD;   // 4,194,304 halves
  _Float16* Q16  = (_Float16*)d_ws;
  _Float16* K16  = Q16 + NHALF;
  _Float16* VT16 = K16 + NHALF;
  float* mArr = (float*)(VT16 + NHALF);
  float* lArr = mArr + (size_t)2 * NH * SEQ;

  const dim3 pgrid(EMB / 64, (2 * SEQ) / 64);       // 16 x 64 tiles of 64x64
  // Q gets the 1/sqrt(head_dim) scale folded in.
  proj_kernel<<<pgrid, 256, 0, stream>>>(problem, Wq, bq, Q16, 0.125f, 0);
  proj_kernel<<<pgrid, 256, 0, stream>>>(context, Wk, bk, K16, 1.0f,   0);
  proj_kernel<<<pgrid, 256, 0, stream>>>(context, Wv, bv, VT16, 1.0f,  1);

  attn_pass1<<<(2 * NH * (SEQ / 16)) / 4, 128, 0, stream>>>(Q16, K16, mArr, lArr);
  attn_pass2<<<(2 * NH * (SEQ / 32)) / 4, 128, 0, stream>>>(Q16, K16, VT16, mArr, lArr,
                                                            maskp, out);
}
